// BinarizedLinearModule_42030549959312
// MI455X (gfx1250) — matrix operand order repaired and checked by reading, not compiled, not run on hardware
//
#include <hip/hip_runtime.h>

// ---------------------------------------------------------------------------
// BinarizedLinear: out[b,o] = sum_i x[b,i] * (u[o,i] < w[o,i])
//   B=16384, K(INUM)=4096, N(ONUM)=512, fp32.
// - Binarized weights are exact in bf16 {0,1.0}; x is split into hi/lo bf16
//   so v_wmma_f32_16x16x32_bf16 (fp32 acc) reproduces fp32-grade results.
// - w/u tiles are double-buffered into LDS with GLOBAL_LOAD_ASYNC_TO_LDS_B128
//   (ASYNCcnt) to hide their latency behind the WMMA stream.
// - Wave tile = 32(M) x 64(N): 16 WMMAs per k-step amortize the binarize,
//   barriers, and B-fragment DS loads.
// ---------------------------------------------------------------------------

typedef __attribute__((ext_vector_type(16))) __bf16 v16bf;
typedef __attribute__((ext_vector_type(8)))  float  v8f;

#define B_DIM 16384
#define K_DIM 4096
#define N_DIM 512
#define BM 256       // M rows per block (8 waves x 32 rows)
#define BN 64        // N cols per block (one wave spans all of it)
#define KT 32        // K per wmma step
#define SB_STRIDE 72 // halves; only sizes sb (its n-rows use stride KT, 64 B)

union FragBF {
  unsigned int u[8];
  uint4 q[2];
  v16bf b;
};

__device__ __forceinline__ void cvt2_hilo(float f0, float f1,
                                          unsigned int& hi, unsigned int& lo) {
  // one dword of bf16 pairs: truncated head + bf16 residual (v_perm packing)
  unsigned int b0 = __float_as_uint(f0);
  unsigned int b1 = __float_as_uint(f1);
  hi = __builtin_amdgcn_perm(b1, b0, 0x07060302u);   // {b1.hi16, b0.hi16}
  float r0 = f0 - __uint_as_float(b0 & 0xFFFF0000u); // exact in fp32
  float r1 = f1 - __uint_as_float(b1 & 0xFFFF0000u);
  lo = __builtin_amdgcn_perm(__float_as_uint(r1), __float_as_uint(r0), 0x07060302u);
}

// One 16-byte async copy per lane: LDS[lds_addr] = *(sbase + goff) (GVS mode)
__device__ __forceinline__ void async_cp_b128(unsigned lds_addr, unsigned goff,
                                              const void* sbase) {
  asm volatile("global_load_async_to_lds_b128 %0, %1, %2"
               :: "v"(lds_addr), "v"(goff), "s"(sbase)
               : "memory");
}

__global__ void __launch_bounds__(256)
binlin_wmma_bf16(const float* __restrict__ x,
                 const float* __restrict__ w,
                 const float* __restrict__ u,
                 float* __restrict__ out) {
  // Double-buffered fp32 staging tiles for w/u: 64 rows x 32 floats (128B rows)
  __shared__ __align__(16) float lsw[2][BN * KT];
  __shared__ __align__(16) float lsu[2][BN * KT];
  // Binarized bf16 tile, layout [n][k], row stride 64 B
  __shared__ __align__(16) unsigned short sb[KT * SB_STRIDE];

  const int tid   = threadIdx.x;
  const int lane  = tid & 31;
  const int wave  = tid >> 5;     // 0..7 -> 32-row M band
  const int lrow  = lane & 15;
  const int lhalf = lane >> 4;

  const int tm = blockIdx.x >> 3; // 64 M tiles
  const int tn = blockIdx.x & 7;  // 8 N tiles
  const int m0 = tm * BM + wave * 32;
  const int n0 = tn * BN;

  // --- async copy chunk mapping: 512 x 16B chunks per 8KB tile, 2 per thread
  const unsigned q0 = tid, q1 = tid + 256;
  const unsigned go0 = (q0 >> 3) * (K_DIM * 4) + (q0 & 7) * 16;
  const unsigned go1 = (q1 >> 3) * (K_DIM * 4) + (q1 & 7) * 16;
  const unsigned ld0 = (q0 >> 3) * (KT * 4) + (q0 & 7) * 16;
  const unsigned ld1 = (q1 >> 3) * (KT * 4) + (q1 & 7) * 16;
  const unsigned lwB = (unsigned)(size_t)&lsw[0][0]; // LDS byte offsets
  const unsigned luB = (unsigned)(size_t)&lsu[0][0];

  const float* wbase = w + (size_t)n0 * K_DIM;
  const float* ubase = u + (size_t)n0 * K_DIM;

  // --- binarize mapping: thread -> (n row, 8-wide k chunk)
  const int bn = tid >> 2;        // 0..63
  const int bk = (tid & 3) * 8;   // 0,8,16,24

  // --- A fragment pointers (ISA A layout: lane<16 -> K off 0/16, lane>=16 -> 8/24)
  const float* arow0 = x + (size_t)(m0 + 0  + lrow) * K_DIM + lhalf * 8;
  const float* arow1 = x + (size_t)(m0 + 16 + lrow) * K_DIM + lhalf * 8;

  v8f acc[2][4];
#pragma unroll
  for (int ms = 0; ms < 2; ++ms)
#pragma unroll
    for (int ns = 0; ns < 4; ++ns)
#pragma unroll
      for (int r = 0; r < 8; ++r) acc[ms][ns][r] = 0.0f;

  // --- prologue: async-stage k=0 tiles into buffer 0
  async_cp_b128(lwB + ld0, go0, wbase);
  async_cp_b128(lwB + ld1, go1, wbase);
  async_cp_b128(luB + ld0, go0, ubase);
  async_cp_b128(luB + ld1, go1, ubase);

  for (int k = 0; k < K_DIM; k += KT) {
    const int buf = (k >> 5) & 1;

    asm volatile("s_wait_asynccnt 0x0" ::: "memory"); // our tile copies landed
    __syncthreads();                                  // everyone's landed

    // ---- binarize 64x32 tile from staged fp32 -> bf16 {0,1} in sb[n][k] ----
    {
      const float4* wv = (const float4*)&lsw[buf][bn * KT + bk];
      const float4* uv = (const float4*)&lsu[buf][bn * KT + bk];
      float4 w0 = wv[0], w1 = wv[1];
      float4 u0 = uv[0], u1 = uv[1];
      unsigned short* sp = &sb[bn * KT + bk];
      sp[0] = (u0.x < w0.x) ? 0x3F80 : 0;
      sp[1] = (u0.y < w0.y) ? 0x3F80 : 0;
      sp[2] = (u0.z < w0.z) ? 0x3F80 : 0;
      sp[3] = (u0.w < w0.w) ? 0x3F80 : 0;
      sp[4] = (u1.x < w1.x) ? 0x3F80 : 0;
      sp[5] = (u1.y < w1.y) ? 0x3F80 : 0;
      sp[6] = (u1.z < w1.z) ? 0x3F80 : 0;
      sp[7] = (u1.w < w1.w) ? 0x3F80 : 0;
    }

    // ---- kick off async staging of the NEXT k tile (other buffer) ----
    if (k + KT < K_DIM) {
      const unsigned bo = (buf ^ 1) * (BN * KT * 4);
      const float* wn2 = wbase + (k + KT);
      const float* un2 = ubase + (k + KT);
      async_cp_b128(lwB + bo + ld0, go0, wn2);
      async_cp_b128(lwB + bo + ld1, go1, wn2);
      async_cp_b128(luB + bo + ld0, go0, un2);
      async_cp_b128(luB + bo + ld1, go1, un2);
    }

    __syncthreads(); // binarized tile visible

    // ---- B fragments: lane holds N=lrow; elems K=8h+0..7, 16+8h+0..7 ----
    FragBF bf[4];
    {
      const char* sbb = (const char*)sb + lrow * (KT * 2) + lhalf * 16;
#pragma unroll
      for (int ns = 0; ns < 4; ++ns) {
        bf[ns].q[0] = *(const uint4*)(sbb + ns * 16 * KT * 2);
        bf[ns].q[1] = *(const uint4*)(sbb + ns * 16 * KT * 2 + 32);
      }
    }

    // ---- per M-subtile: A fragment (hi/lo bf16) + 8 WMMAs ----
#pragma unroll
    for (int ms = 0; ms < 2; ++ms) {
      const float* ap = (ms ? arow1 : arow0) + k;
      __builtin_prefetch(ap + KT, 0, 1); // global_prefetch_b8: next K tile of x
      float4 a0 = ((const float4*)ap)[0];
      float4 a1 = ((const float4*)ap)[1];
      float4 a2 = ((const float4*)(ap + 16))[0];
      float4 a3 = ((const float4*)(ap + 16))[1];

      FragBF ahi, alo;
      cvt2_hilo(a0.x, a0.y, ahi.u[0], alo.u[0]);
      cvt2_hilo(a0.z, a0.w, ahi.u[1], alo.u[1]);
      cvt2_hilo(a1.x, a1.y, ahi.u[2], alo.u[2]);
      cvt2_hilo(a1.z, a1.w, ahi.u[3], alo.u[3]);
      cvt2_hilo(a2.x, a2.y, ahi.u[4], alo.u[4]);
      cvt2_hilo(a2.z, a2.w, ahi.u[5], alo.u[5]);
      cvt2_hilo(a3.x, a3.y, ahi.u[6], alo.u[6]);
      cvt2_hilo(a3.z, a3.w, ahi.u[7], alo.u[7]);

#pragma unroll
      for (int ns = 0; ns < 4; ++ns) {
        acc[ms][ns] = __builtin_amdgcn_wmma_f32_16x16x32_bf16(
            false, ahi.b, false, bf[ns].b, (short)0, acc[ms][ns], false, false);
        acc[ms][ns] = __builtin_amdgcn_wmma_f32_16x16x32_bf16(
            false, alo.b, false, bf[ns].b, (short)0, acc[ms][ns], false, false);
      }
    }
  }

  // ---- store C: VGPR r -> M = r + 8*lhalf, N = lrow (ISA C/D layout) ----
#pragma unroll
  for (int ms = 0; ms < 2; ++ms) {
#pragma unroll
    for (int ns = 0; ns < 4; ++ns) {
      const int col = n0 + ns * 16 + lrow;
#pragma unroll
      for (int r = 0; r < 8; ++r) {
        const int row = m0 + ms * 16 + lhalf * 8 + r;
        out[(size_t)row * N_DIM + col] = acc[ms][ns][r];
      }
    }
  }
}

extern "C" void kernel_launch(void* const* d_in, const int* in_sizes, int n_in,
                              void* d_out, int out_size, void* d_ws, size_t ws_size,
                              hipStream_t stream) {
  const float* x = (const float*)d_in[0];   // [16384, 4096]
  const float* w = (const float*)d_in[1];   // [512, 4096]
  const float* u = (const float*)d_in[2];   // [512, 4096]
  float* out = (float*)d_out;               // [16384, 512]

  dim3 grid((B_DIM / BM) * (N_DIM / BN));   // 64 * 8 = 512 blocks
  dim3 block(256);                          // 8 wave32
  binlin_wmma_bf16<<<grid, block, 0, stream>>>(x, w, u, out);
}
